// ContrastiveLoss_23553600651798
// MI455X (gfx1250) — matrix operand order repaired and checked by reading, not compiled, not run on hardware
//
#include <hip/hip_runtime.h>
#include <hip/hip_bf16.h>

typedef _Float16 v16h __attribute__((ext_vector_type(16)));
typedef _Float16 v8h  __attribute__((ext_vector_type(8)));
typedef _Float16 v4h  __attribute__((ext_vector_type(4)));
typedef float    v8f  __attribute__((ext_vector_type(8)));
typedef float    v4f  __attribute__((ext_vector_type(4)));

#define BB    8192
#define DD    128
#define NCLS  64
#define NSPL  8                 // column splits per row-tile
#define INV_T 14.285714285714286f   // 1/0.07 ; also the fixed LSE shift C

// ---------------------------------------------------------------- workspace
// [0, 2MB)        : Ehi  (8192x128 f16)
// [2MB, 4MB)      : Elo  (8192x128 f16)
// [4MB, +256KB)   : s_part  (NSPL x 8192 f32)  -- per-split partial exp sums
// then            : s_row   (8192 f32)
// then            : loss_part (4096 f32)
// then            : hist    (64 u32)
#define WS_EHI   0
#define WS_ELO   (2u * 1024u * 1024u)
#define WS_SPART (4u * 1024u * 1024u)
#define WS_SROW  (WS_SPART + NSPL * BB * 4u)
#define WS_LPART (WS_SROW + BB * 4u)
#define WS_HIST  (WS_LPART + (BB / 16) * NSPL * 4u)

// ---------------------------------------------------------------- init + hist
__global__ void k_init(unsigned int* __restrict__ hist) {
  int t = threadIdx.x;
  if (t < NCLS) hist[t] = 0u;
}

__global__ void k_hist(const int* __restrict__ labels, unsigned int* __restrict__ hist) {
  int i = blockIdx.x * blockDim.x + threadIdx.x;
  if (i < BB) atomicAdd(&hist[labels[i] & (NCLS - 1)], 1u);
}

// ---------------------------------------------------------------- normalize + hi/lo split
// one wave (32 lanes) per row; lane handles 4 contiguous floats
__global__ void __launch_bounds__(256) k_normalize(const float* __restrict__ x,
                                                   _Float16* __restrict__ ehi,
                                                   _Float16* __restrict__ elo) {
  int row  = (blockIdx.x * blockDim.x + threadIdx.x) >> 5;
  int lane = threadIdx.x & 31;
  if (row >= BB) return;
  v4f v = *(const v4f*)(x + (size_t)row * DD + lane * 4);
  float ss = v.x * v.x + v.y * v.y + v.z * v.z + v.w * v.w;
  #pragma unroll
  for (int o = 16; o >= 1; o >>= 1) ss += __shfl_xor(ss, o, 32);
  float rn = 1.0f / fmaxf(sqrtf(ss), 1e-12f);
  v4h hi, lo;
  #pragma unroll
  for (int c = 0; c < 4; ++c) {
    float e = v[c] * rn;
    _Float16 h = (_Float16)e;
    hi[c] = h;
    lo[c] = (_Float16)(e - (float)h);
  }
  *(v4h*)(ehi + (size_t)row * DD + lane * 4) = hi;
  *(v4h*)(elo + (size_t)row * DD + lane * 4) = lo;
}

// ---------------------------------------------------------------- fragment loads (ISA layouts)
// A 16x32 f16: lane (g = lane>>4, m = lane&15) row = m;
//   halves 0..7  = K kc + 8g .. +7 ; halves 8..15 = K kc + 16 + 8g .. +7
__device__ __forceinline__ v16h load_a_frag(const _Float16* __restrict__ e,
                                            int row, int kc, int g) {
  const _Float16* p = e + (size_t)row * DD + kc + 8 * g;
  v8h a = *(const v8h*)p;
  v8h b = *(const v8h*)(p + 16);
  v16h r;
  #pragma unroll
  for (int t = 0; t < 8; ++t) { r[t] = a[t]; r[t + 8] = b[t]; }
  return r;
}
// B 32x16 f16: lane col = m; halves 0..7 = K kc + 8g .. +7 ; halves 8..15 = K kc + 16 + 8g .. +7
__device__ __forceinline__ v16h load_b_frag(const _Float16* __restrict__ e,
                                            int col, int kc, int g) {
  const _Float16* p = e + (size_t)col * DD + kc + 8 * g;
  v8h a = *(const v8h*)p;
  v8h b = *(const v8h*)(p + 16);
  v16h r;
  #pragma unroll
  for (int t = 0; t < 8; ++t) { r[t] = a[t]; r[t + 8] = b[t]; }
  return r;
}

// 16x16x128 hi/lo-compensated Gram tile: C += Ahi*Bhi + Ahi*Blo + Alo*Bhi
__device__ __forceinline__ v8f gram_tile(const v16h* __restrict__ ahi,
                                         const v16h* __restrict__ alo,
                                         const _Float16* __restrict__ ehi,
                                         const _Float16* __restrict__ elo,
                                         int j0, int m, int g) {
  v8f c = {};
  #pragma unroll
  for (int k = 0; k < 4; ++k) {
    v16h bhi = load_b_frag(ehi, j0 + m, 32 * k, g);
    v16h blo = load_b_frag(elo, j0 + m, 32 * k, g);
    c = __builtin_amdgcn_wmma_f32_16x16x32_f16(false, ahi[k], false, bhi,
                                               (short)0, c, false, false);
    c = __builtin_amdgcn_wmma_f32_16x16x32_f16(false, ahi[k], false, blo,
                                               (short)0, c, false, false);
    c = __builtin_amdgcn_wmma_f32_16x16x32_f16(false, alo[k], false, bhi,
                                               (short)0, c, false, false);
  }
  return c;
}

// ---------------------------------------------------------------- pass 2: masked exp-sums
// grid (BB/16, NSPL), block 32. Wave owns 16 rows x 1024 columns.
// s_part[split][row] = sum over negatives in column range of exp(sim - C)
__global__ void __launch_bounds__(32) k_neg_sum(const _Float16* __restrict__ ehi,
                                                const _Float16* __restrict__ elo,
                                                const int* __restrict__ labels,
                                                float* __restrict__ s_part) {
  const int i0   = blockIdx.x * 16;
  const int lane = threadIdx.x;
  const int g = lane >> 4, m = lane & 15;

  v16h ahi[4], alo[4];
  #pragma unroll
  for (int k = 0; k < 4; ++k) {
    ahi[k] = load_a_frag(ehi, i0 + m, 32 * k, g);
    alo[k] = load_a_frag(elo, i0 + m, 32 * k, g);
  }
  int rlab[8];
  #pragma unroll
  for (int r = 0; r < 8; ++r) rlab[r] = labels[i0 + 8 * g + r];

  float s[8];
  #pragma unroll
  for (int r = 0; r < 8; ++r) s[r] = 0.0f;

  const int jbeg = blockIdx.y * (BB / NSPL);
  const int jend = jbeg + (BB / NSPL);
  for (int j0 = jbeg; j0 < jend; j0 += 16) {
    __builtin_prefetch(ehi + (size_t)(j0 + 16 + m) * DD, 0, 1);  // global_prefetch_b8
    v8f c = gram_tile(ahi, alo, ehi, elo, j0, m, g);
    int jlab = labels[j0 + m];
    #pragma unroll
    for (int r = 0; r < 8; ++r) {
      float xv = c[r] * INV_T;                 // sim
      float ex = __expf(xv - INV_T);           // exp(sim - C), never overflows
      s[r] += (jlab != rlab[r]) ? ex : 0.0f;   // negatives only (diag is same-label)
    }
  }
  // row sums live strided across the 16 lanes of each half-wave group
  #pragma unroll
  for (int r = 0; r < 8; ++r) {
    float t = s[r];
    #pragma unroll
    for (int o = 8; o >= 1; o >>= 1) t += __shfl_xor(t, o, 16);
    if (m == 0) s_part[(size_t)blockIdx.y * BB + i0 + 8 * g + r] = t;
  }
}

// fixed-order reduce of the NSPL partials -> s_row (deterministic)
__global__ void k_sreduce(const float* __restrict__ s_part, float* __restrict__ s_row) {
  int i = blockIdx.x * blockDim.x + threadIdx.x;
  if (i >= BB) return;
  float t = 0.0f;
  #pragma unroll
  for (int p = 0; p < NSPL; ++p) t += s_part[(size_t)p * BB + i];
  s_row[i] = t;
}

// ---------------------------------------------------------------- pass 3: positive-pair loss
// pair_loss = log1p(s_row[i] * exp(C - sim[i,j])) over valid positive pairs
__global__ void __launch_bounds__(32) k_pos_loss(const _Float16* __restrict__ ehi,
                                                 const _Float16* __restrict__ elo,
                                                 const int* __restrict__ labels,
                                                 const float* __restrict__ s_row,
                                                 const unsigned int* __restrict__ hist,
                                                 float* __restrict__ loss_part) {
  const int i0   = blockIdx.x * 16;
  const int lane = threadIdx.x;
  const int g = lane >> 4, m = lane & 15;

  v16h ahi[4], alo[4];
  #pragma unroll
  for (int k = 0; k < 4; ++k) {
    ahi[k] = load_a_frag(ehi, i0 + m, 32 * k, g);
    alo[k] = load_a_frag(elo, i0 + m, 32 * k, g);
  }
  int rlab[8]; float rs[8]; int rvalid[8];
  #pragma unroll
  for (int r = 0; r < 8; ++r) {
    int row = i0 + 8 * g + r;
    rlab[r] = labels[row];
    rs[r]   = s_row[row];
    unsigned int h = hist[rlab[r] & (NCLS - 1)];
    rvalid[r] = (h >= 2u) && (h < (unsigned)BB);   // has a positive and a negative
  }

  float acc = 0.0f;
  const int jbeg = blockIdx.y * (BB / NSPL);
  const int jend = jbeg + (BB / NSPL);
  for (int j0 = jbeg; j0 < jend; j0 += 16) {
    __builtin_prefetch(ehi + (size_t)(j0 + 16 + m) * DD, 0, 1);
    v8f c = gram_tile(ahi, alo, ehi, elo, j0, m, g);
    int jlab = labels[j0 + m];
    #pragma unroll
    for (int r = 0; r < 8; ++r) {
      int row = i0 + 8 * g + r;
      bool pos = (jlab == rlab[r]) && ((j0 + m) != row) && rvalid[r];
      if (pos) {
        float xv = c[r] * INV_T;
        acc += log1pf(rs[r] * __expf(INV_T - xv));
      }
    }
  }
  #pragma unroll
  for (int o = 16; o >= 1; o >>= 1) acc += __shfl_xor(acc, o, 32);
  if (lane == 0) loss_part[(size_t)blockIdx.y * gridDim.x + blockIdx.x] = acc;
}

// ---------------------------------------------------------------- finalize (1 wave)
__global__ void __launch_bounds__(32) k_finalize(const float* __restrict__ loss_part,
                                                 const unsigned int* __restrict__ hist,
                                                 float* __restrict__ out) {
  int lane = threadIdx.x;
  float ls = 0.0f;
  for (int t = lane; t < (BB / 16) * NSPL; t += 32) ls += loss_part[t];
  #pragma unroll
  for (int o = 16; o >= 1; o >>= 1) ls += __shfl_xor(ls, o, 32);

  unsigned int cnt = 0;
  for (int c = lane; c < NCLS; c += 32) {
    unsigned int h = hist[c];
    if (h >= 2u && h < (unsigned)BB) cnt += h * (h - 1u);  // valid positive pairs
  }
  #pragma unroll
  for (int o = 16; o >= 1; o >>= 1) cnt += __shfl_xor((int)cnt, o, 32);

  if (lane == 0) out[0] = (cnt > 0u) ? (ls / (float)cnt) : 0.0f;
}

// ---------------------------------------------------------------- launcher
extern "C" void kernel_launch(void* const* d_in, const int* in_sizes, int n_in,
                              void* d_out, int out_size, void* d_ws, size_t ws_size,
                              hipStream_t stream) {
  const float* emb   = (const float*)d_in[0];
  const int*   labels = (const int*)d_in[1];
  float* out = (float*)d_out;

  char* ws = (char*)d_ws;
  _Float16*     ehi    = (_Float16*)(ws + WS_EHI);
  _Float16*     elo    = (_Float16*)(ws + WS_ELO);
  float*        s_part = (float*)(ws + WS_SPART);
  float*        s_row  = (float*)(ws + WS_SROW);
  float*        l_part = (float*)(ws + WS_LPART);
  unsigned int* hist   = (unsigned int*)(ws + WS_HIST);

  k_init<<<1, 64, 0, stream>>>(hist);
  k_hist<<<BB / 256, 256, 0, stream>>>(labels, hist);
  k_normalize<<<BB / 8, 256, 0, stream>>>(emb, ehi, elo);

  dim3 tgrid(BB / 16, NSPL);
  k_neg_sum<<<tgrid, 32, 0, stream>>>(ehi, elo, labels, s_part);
  k_sreduce<<<BB / 256, 256, 0, stream>>>(s_part, s_row);
  k_pos_loss<<<tgrid, 32, 0, stream>>>(ehi, elo, labels, s_row, hist, l_part);
  k_finalize<<<1, 32, 0, stream>>>(l_part, hist, out);
}
